// GraphConv_23459111371106
// MI455X (gfx1250) — matrix operand order repaired and checked by reading, not compiled, not run on hardware
//
#include <hip/hip_runtime.h>
#include <hip/hip_bf16.h>
#include <math.h>

// ---------------------------------------------------------------- constants
#define Bb   64
#define Nn   36
#define Dd   2048
#define Kk   8
#define Mm   128
#define ROWS (Bb * Nn)          // 2304 GEMM rows (b,j)
#define COLS (Kk * Mm)          // 1024 GEMM cols (k,m)

typedef __attribute__((ext_vector_type(16))) __bf16 v16bf;
typedef __attribute__((ext_vector_type(8)))  __bf16 v8bf;
typedef __attribute__((ext_vector_type(8)))  float  v8f;

static __device__ __forceinline__ v16bf cat16(v8bf x, v8bf y) {
    v16bf r;
#pragma unroll
    for (int i = 0; i < 8; ++i) { r[i] = x[i]; r[i + 8] = y[i]; }
    return r;
}

#define WMMA_BF16(Afrag, Bfrag, Cacc) \
    __builtin_amdgcn_wmma_f32_16x16x32_bf16(false, (Afrag), false, (Bfrag), \
                                            (short)0, (Cacc), false, false)

// ------------------------------------------------- stage A: gaussian weights
// W[b,k,i,j] = normalized_gaussian(b,i,j,k) * adj_matrix[b,i,j] * adj[b,i,j]
__global__ void gc_weights(const float* __restrict__ coord,
                           const float* __restrict__ adjm,
                           const float* __restrict__ adj,
                           const float* __restrict__ mr,
                           const float* __restrict__ mt,
                           const float* __restrict__ pr,
                           const float* __restrict__ pt,
                           float* __restrict__ W) {
    int idx = blockIdx.x * blockDim.x + threadIdx.x;     // over B*N*N
    if (idx >= Bb * Nn * Nn) return;
    int j = idx % Nn;
    int i = (idx / Nn) % Nn;
    int b = idx / (Nn * Nn);

    float rho = coord[2 * idx + 0];
    float th  = coord[2 * idx + 1];
    float a   = adjm[idx] * adj[idx];

    const float TWO_PI = 6.28318530717958647692f;
    float w[Kk];
    float s = 0.0f;
#pragma unroll
    for (int k = 0; k < Kk; ++k) {
        float dr   = rho - mr[k];
        float prk  = pr[k];
        float wr   = __expf(-0.5f * dr * dr / (1e-14f + prk * prk));
        float fa   = fabsf(th - mt[k]);
        float sa   = fabsf(TWO_PI - fa);
        float da   = fminf(fa, sa);
        float ptk  = pt[k];
        float wt   = __expf(-0.5f * da * da / (1e-14f + ptk * ptk));
        float ww   = wr * wt;
        ww = (ww != ww) ? 0.0f : ww;                     // NaN -> 0
        w[k] = ww;
        s += ww;
    }
    float scale = a / (s + 1e-14f);
#pragma unroll
    for (int k = 0; k < Kk; ++k)
        W[((size_t)(b * Kk + k) * Nn + i) * Nn + j] = w[k] * scale;
}

// ----------------- stage B0: split v (fp32 -> bf16 hi/lo, interleaved rows)
// Layout: vhl[(row*2 + 0)*Dd + d] = hi, vhl[(row*2 + 1)*Dd + d] = lo
__global__ void gc_split_v(const float* __restrict__ v,
                           __bf16* __restrict__ vhl) {
    int idx = blockIdx.x * blockDim.x + threadIdx.x;     // over ROWS*D
    if (idx >= ROWS * Dd) return;
    int d   = idx % Dd;
    int row = idx / Dd;
    float x  = v[idx];
    __bf16 h = (__bf16)x;
    size_t o = ((size_t)row * 2) * Dd + d;
    vhl[o]      = h;
    vhl[o + Dd] = (__bf16)(x - (float)h);
}

// --- stage B1: split+transpose conv_w[K,D,M] -> whl[(k*M+m)*2][D] (hi|lo rows)
__global__ void gc_split_w(const float* __restrict__ cw,
                           __bf16* __restrict__ whl) {
    int idx = blockIdx.x * blockDim.x + threadIdx.x;     // over K*D*M
    if (idx >= Kk * Dd * Mm) return;
    int m = idx % Mm;
    int d = (idx / Mm) % Dd;
    int k = idx / (Mm * Dd);
    float x  = cw[idx];
    __bf16 h = (__bf16)x;
    size_t o = ((size_t)(k * Mm + m) * 2) * Dd + d;
    whl[o]      = h;
    whl[o + Dd] = (__bf16)(x - (float)h);
}

// ---------------- stage B2: VC[row, col] = v[row,:] @ conv_wT[col,:] (bf16x3)
// One wave computes a 32x64 tile (2 row-frags x 4 col-tiles, 8 accumulators).
// All 24 loads per k-step address off TWO base pointers with immediate
// offsets (hi/lo interleaving keeps every offset within the 24-bit IOFFSET).
// __launch_bounds__(128, 1): relax occupancy so the allocator keeps both
// base-pointer pairs resident (no scratch spill / loadcnt-drain in the loop).
__global__ void __launch_bounds__(128, 1)
gc_gemm_vc(const __bf16* __restrict__ vhl,
           const __bf16* __restrict__ whl,
           float* __restrict__ vc) {
    const int lane    = threadIdx.x & 31;
    const int wave    = threadIdx.x >> 5;
    const int g       = lane >> 4;          // lane half: 0 or 1
    const int rc      = lane & 15;          // row (A) / col (B) within tile
    const int rowBase = blockIdx.x * 32;                 // 2 row tiles
    const int colBase = blockIdx.y * 256 + wave * 64;    // 4 col tiles

    // A element mapping: e=0..7 -> K=kk+8g+e ; e=8..15 -> K=kk+16+8g+(e-8)
    const __bf16* aB = vhl + ((size_t)(rowBase + rc) * 2) * Dd + 8 * g;
    // B element mapping: e=0..7 -> K=kk+8g+e ; e=8..15 -> K=kk+16+8g+(e-8)
    const __bf16* bB = whl + ((size_t)(colBase + rc) * 2) * Dd + 8 * g;

    v8f acc[2][4] = {};
    for (int kk = 0; kk < Dd; kk += 32) {
        v16bf ahi0 = cat16(*(const v8bf*)(aB),
                           *(const v8bf*)(aB + 16));
        v16bf alo0 = cat16(*(const v8bf*)(aB + Dd),
                           *(const v8bf*)(aB + Dd + 16));
        v16bf ahi1 = cat16(*(const v8bf*)(aB + 32 * Dd),
                           *(const v8bf*)(aB + 32 * Dd + 16));
        v16bf alo1 = cat16(*(const v8bf*)(aB + 33 * Dd),
                           *(const v8bf*)(aB + 33 * Dd + 16));
#pragma unroll
        for (int t = 0; t < 4; ++t) {
            const __bf16* bt = bB + (size_t)t * 32 * Dd;
            v16bf bhi = cat16(*(const v8bf*)(bt),
                              *(const v8bf*)(bt + 16));
            v16bf blo = cat16(*(const v8bf*)(bt + Dd),
                              *(const v8bf*)(bt + Dd + 16));
            // bf16x3; adjacent WMMAs alternate accumulators (no C-RAW pairs)
            acc[0][t] = WMMA_BF16(ahi0, bhi, acc[0][t]);
            acc[1][t] = WMMA_BF16(ahi1, bhi, acc[1][t]);
            acc[0][t] = WMMA_BF16(ahi0, blo, acc[0][t]);
            acc[1][t] = WMMA_BF16(ahi1, blo, acc[1][t]);
            acc[0][t] = WMMA_BF16(alo0, bhi, acc[0][t]);
            acc[1][t] = WMMA_BF16(alo1, bhi, acc[1][t]);
        }
        aB += 32;
        bB += 32;
    }
    // D layout: vgpr r -> M = r + 8g ; N = rc
#pragma unroll
    for (int rt = 0; rt < 2; ++rt)
#pragma unroll
        for (int t = 0; t < 4; ++t)
#pragma unroll
            for (int r = 0; r < 8; ++r)
                vc[(size_t)(rowBase + 16 * rt + r + 8 * g) * COLS
                   + colBase + 16 * t + rc] = acc[rt][t][r];
}

// --------------- stage C: out[b,i,c] = sum_j W[b,k,i,j] * VC[(b,j), c], c=(k,m)
__global__ void gc_epilogue(const float* __restrict__ W,
                            const float* __restrict__ vc,
                            float* __restrict__ out) {
    int idx = blockIdx.x * blockDim.x + threadIdx.x;     // over ROWS*COLS
    if (idx >= ROWS * COLS) return;
    int c   = idx & (COLS - 1);
    int row = idx >> 10;                                  // b*N + i
    int b   = row / Nn;
    int i   = row % Nn;
    int k   = c >> 7;

    const float* wrow = W  + ((size_t)(b * Kk + k) * Nn + i) * Nn;
    const float* vcol = vc + (size_t)b * Nn * COLS + c;
    float acc = 0.0f;
#pragma unroll
    for (int j = 0; j < Nn; ++j)
        acc = fmaf(wrow[j], vcol[(size_t)j * COLS], acc);
    out[idx] = acc;
}

// ---------------------------------------------------------------- launcher
extern "C" void kernel_launch(void* const* d_in, const int* in_sizes, int n_in,
                              void* d_out, int out_size, void* d_ws, size_t ws_size,
                              hipStream_t stream) {
    const float* v     = (const float*)d_in[0];
    // d_in[1] = v_mask (unused by reference)
    const float* coord = (const float*)d_in[2];
    const float* adjm  = (const float*)d_in[3];
    const float* adj   = (const float*)d_in[4];
    // d_in[5] = top_ind (unused by reference)
    const float* mr    = (const float*)d_in[6];
    const float* mt    = (const float*)d_in[7];
    const float* pr    = (const float*)d_in[8];
    const float* pt    = (const float*)d_in[9];
    const float* convw = (const float*)d_in[10];
    float* out = (float*)d_out;

    char*  ws  = (char*)d_ws;
    size_t off = 0;
    auto carve = [&](size_t bytes) -> void* {
        void* p = ws + off;
        off += (bytes + 255) & ~(size_t)255;
        return p;
    };
    float*  W   = (float*)carve((size_t)Bb * Kk * Nn * Nn * sizeof(float));
    __bf16* vhl = (__bf16*)carve((size_t)ROWS * 2 * Dd * sizeof(__bf16));
    __bf16* whl = (__bf16*)carve((size_t)COLS * 2 * Dd * sizeof(__bf16));
    float*  vc  = (float*)carve((size_t)ROWS * COLS * sizeof(float));
    (void)ws_size; (void)in_sizes; (void)n_in; (void)out_size;

    {   // stage A
        int n = Bb * Nn * Nn;
        gc_weights<<<(n + 255) / 256, 256, 0, stream>>>(coord, adjm, adj,
                                                        mr, mt, pr, pt, W);
    }
    {   // stage B0
        int n = ROWS * Dd;
        gc_split_v<<<(n + 255) / 256, 256, 0, stream>>>(v, vhl);
    }
    {   // stage B1
        int n = Kk * Dd * Mm;
        gc_split_w<<<(n + 255) / 256, 256, 0, stream>>>(convw, whl);
    }
    {   // stage B2: 72 row tiles x 4 col groups; 128 threads = 4 waves
        dim3 grid(ROWS / 32, COLS / 256);
        gc_gemm_vc<<<grid, 128, 0, stream>>>(vhl, whl, vc);
    }
    {   // stage C
        int n = ROWS * COLS;
        gc_epilogue<<<(n + 255) / 256, 256, 0, stream>>>(W, vc, out);
    }
}
